// AttentionVisual_65094524338972
// MI455X (gfx1250) — matrix operand order repaired and checked by reading, not compiled, not run on hardware
//
#include <hip/hip_runtime.h>
#include <math.h>

// ---------------------------------------------------------------------------
// Problem dims (compile-time constants, match reference)
// ---------------------------------------------------------------------------
static constexpr int NB = 512;    // batch
static constexpr int NR = 196;    // regions
static constexpr int DV = 1024;   // visual dim (K of big GEMM)
static constexpr int DH = 512;    // hidden dim
static constexpr int DA = 256;    // attention dim (N of big GEMM)

static constexpr int MT = 13;     // ceil(196/16) M-tiles per batch
static constexpr int KSTEPS = DV / 32;   // 32 bf16-wmma K-steps

typedef __attribute__((ext_vector_type(16))) __bf16 v16bf;
typedef __attribute__((ext_vector_type(8)))  float  v8f;

union BF16x16 { unsigned int u[8]; v16bf v; };

__device__ __forceinline__ unsigned int pk_bf16(float lo, float hi) {
    // element 2j -> bits[15:0], element 2j+1 -> bits[31:16]; bf16 via truncation
    return (__float_as_uint(lo) >> 16) | (__float_as_uint(hi) & 0xffff0000u);
}
__device__ __forceinline__ float bf_resid(float x) {
    // residual after bf16 truncation (x - bf16(x)), itself stored as bf16
    return x - __uint_as_float(__float_as_uint(x) & 0xffff0000u);
}
__device__ __forceinline__ v8f wmma_bf16(v16bf a, v16bf b, v8f c) {
    return __builtin_amdgcn_wmma_f32_16x16x32_bf16(false, a, false, b,
                                                   (short)0, c, false, false);
}

// ---------------------------------------------------------------------------
// K0: pre-swizzle Wv (f32 [DV][DA] row-major) into WMMA-B-fragment-ready
//     bf16 hi/lo buffers. Fragment (ks, nt): lane half h holds K = ks*32+h*16+e
//     for e=0..15 (2 per dword), column n = nt*16 + (lane&15).
//     word index = ((ks*16 + nt)*32 + lane)*8 + j, elements (2j, 2j+1).
// ---------------------------------------------------------------------------
__global__ void k0_swizzle_wv(const float* __restrict__ Wv,
                              unsigned int* __restrict__ bhi,
                              unsigned int* __restrict__ blo) {
    int tid  = blockIdx.x * blockDim.x + threadIdx.x;   // 0 .. 131071
    int j    = tid & 7;
    int lane = (tid >> 3) & 31;
    int nt   = (tid >> 8) & 15;
    int ks   = tid >> 12;
    int half = lane >> 4;
    int n    = nt * 16 + (lane & 15);
    int k0   = ks * 32 + half * 16 + 2 * j;
    float w0 = Wv[(size_t)k0 * DA + n];
    float w1 = Wv[(size_t)(k0 + 1) * DA + n];
    bhi[tid] = pk_bf16(w0, w1);
    blo[tid] = pk_bf16(bf_resid(w0), bf_resid(w1));
}

// ---------------------------------------------------------------------------
// K1: h_proj[b][n] = hidden[b] @ Wh + bh   (tiny: 0.13 GFLOP, Wh L2-resident)
// ---------------------------------------------------------------------------
__global__ void k1_hproj(const float* __restrict__ hidden,
                         const float* __restrict__ Wh,
                         const float* __restrict__ bh,
                         float* __restrict__ hp) {
    int b = blockIdx.x;
    int n = threadIdx.x;                 // 0..255
    float acc = bh[n];
    const float* hrow = hidden + (size_t)b * DH;
    #pragma unroll 4
    for (int k = 0; k < DH; ++k)
        acc = fmaf(hrow[k], Wh[(size_t)k * DA + n], acc);
    hp[(size_t)b * DA + n] = acc;
}

// ---------------------------------------------------------------------------
// K2: scores via split-precision bf16 WMMA.
//     block = (m-tile, batch); 128 threads = 4 waves; wave w owns N-tiles
//     w*4 .. w*4+3. C = (Ah+Al)(Bh+Bl) ~= Ah*Bh + Ah*Bl + Al*Bh (f32 accum,
//     ~2^-16 relative accuracy). Then joint = relu(C + bv + hproj);
//     scores[r] = joint . Wa + ba.
// ---------------------------------------------------------------------------
__global__ void __launch_bounds__(128)
k2_scores(const float* __restrict__ vf,
          const unsigned int* __restrict__ bhi,
          const unsigned int* __restrict__ blo,
          const float* __restrict__ hp,
          const float* __restrict__ bv,
          const float* __restrict__ Wa,
          const float* __restrict__ ba_p,
          float* __restrict__ scores) {
    const int mt = blockIdx.x;
    const int b  = blockIdx.y;
    const int r0 = mt * 16;

    __shared__ float s_sc[16];
    if (threadIdx.x < 16) s_sc[threadIdx.x] = 0.0f;
    __syncthreads();

    const int wave = threadIdx.x >> 5;
    const int lane = threadIdx.x & 31;
    const int half = lane >> 4;
    const int lsub = lane & 15;

    // A-fragment row for this lane (16-bit A layout: both halves carry M=0..15)
    const int r = r0 + lsub;
    const bool rvalid = (r < NR);
    const float* arow  = vf + ((size_t)b * NR + r) * DV;
    // clamped row pointer for speculative prefetch (always in-bounds)
    const int rc = (r < NR) ? r : (NR - 1);
    const float* prow  = vf + ((size_t)b * NR + rc) * DV;

    v8f acc[4] = {};   // 4 N-tiles per wave, f32 16x16 accumulators

    #pragma unroll 2
    for (int ks = 0; ks < KSTEPS; ++ks) {
        // prefetch the A-chunk two K-steps ahead (HBM stream of this kernel)
        if (ks + 2 < KSTEPS) {
            __builtin_prefetch(prow + (ks + 2) * 32 + half * 8, 0, 3);
        }

        // ---- A fragment: lanes 0-15 hold K = ks*32 + {0..15},
        //                  lanes 16-31 hold K = ks*32 + {16..31}
        const int k1 = ks * 32 + half * 16;
        const int k2 = k1 + 8;
        float4 a00 = {0,0,0,0}, a01 = {0,0,0,0}, a10 = {0,0,0,0}, a11 = {0,0,0,0};
        if (rvalid) {
            a00 = *(const float4*)(arow + k1);
            a01 = *(const float4*)(arow + k1 + 4);
            a10 = *(const float4*)(arow + k2);
            a11 = *(const float4*)(arow + k2 + 4);
        }
        BF16x16 ah, al;
        ah.u[0] = pk_bf16(a00.x, a00.y); ah.u[1] = pk_bf16(a00.z, a00.w);
        ah.u[2] = pk_bf16(a01.x, a01.y); ah.u[3] = pk_bf16(a01.z, a01.w);
        ah.u[4] = pk_bf16(a10.x, a10.y); ah.u[5] = pk_bf16(a10.z, a10.w);
        ah.u[6] = pk_bf16(a11.x, a11.y); ah.u[7] = pk_bf16(a11.z, a11.w);
        al.u[0] = pk_bf16(bf_resid(a00.x), bf_resid(a00.y));
        al.u[1] = pk_bf16(bf_resid(a00.z), bf_resid(a00.w));
        al.u[2] = pk_bf16(bf_resid(a01.x), bf_resid(a01.y));
        al.u[3] = pk_bf16(bf_resid(a01.z), bf_resid(a01.w));
        al.u[4] = pk_bf16(bf_resid(a10.x), bf_resid(a10.y));
        al.u[5] = pk_bf16(bf_resid(a10.z), bf_resid(a10.w));
        al.u[6] = pk_bf16(bf_resid(a11.x), bf_resid(a11.y));
        al.u[7] = pk_bf16(bf_resid(a11.z), bf_resid(a11.w));

        #pragma unroll
        for (int i = 0; i < 4; ++i) {
            const int nt = wave * 4 + i;
            const size_t base = ((size_t)(ks * 16 + nt) * 32 + lane) * 8;
            uint4 h0 = *(const uint4*)(bhi + base);
            uint4 h1 = *(const uint4*)(bhi + base + 4);
            uint4 l0 = *(const uint4*)(blo + base);
            uint4 l1 = *(const uint4*)(blo + base + 4);
            BF16x16 bh_, bl_;
            bh_.u[0]=h0.x; bh_.u[1]=h0.y; bh_.u[2]=h0.z; bh_.u[3]=h0.w;
            bh_.u[4]=h1.x; bh_.u[5]=h1.y; bh_.u[6]=h1.z; bh_.u[7]=h1.w;
            bl_.u[0]=l0.x; bl_.u[1]=l0.y; bl_.u[2]=l0.z; bl_.u[3]=l0.w;
            bl_.u[4]=l1.x; bl_.u[5]=l1.y; bl_.u[6]=l1.z; bl_.u[7]=l1.w;

            acc[i] = wmma_bf16(ah.v, bh_.v, acc[i]);  // Ah*Bh
            acc[i] = wmma_bf16(ah.v, bl_.v, acc[i]);  // Ah*Bl
            acc[i] = wmma_bf16(al.v, bh_.v, acc[i]);  // Al*Bh
        }
    }

    // ---- epilogue: bias + relu + dot(Wa), reduce across columns.
    // C layout: lanes 0-15 -> (row v, col lsub); lanes 16-31 -> (row v+8, col lsub)
    #pragma unroll
    for (int i = 0; i < 4; ++i) {
        const int n = (wave * 4 + i) * 16 + lsub;
        const float bias = bv[n] + hp[(size_t)b * DA + n];
        const float wa   = Wa[n];
        #pragma unroll
        for (int v = 0; v < 8; ++v) {
            float val = acc[i][v] + bias;
            val = fmaxf(val, 0.0f);
            float t = val * wa;
            // reduce over the 16 lanes of each half (columns of this N-tile)
            t += __shfl_xor(t, 1, 32);
            t += __shfl_xor(t, 2, 32);
            t += __shfl_xor(t, 4, 32);
            t += __shfl_xor(t, 8, 32);
            if (lsub == 0)
                atomicAdd(&s_sc[half * 8 + v], t);   // ds_add_f32
        }
    }
    __syncthreads();

    if (threadIdx.x < 16) {
        int rr = r0 + threadIdx.x;
        if (rr < NR)
            scores[(size_t)b * NR + rr] = s_sc[threadIdx.x] + ba_p[0];
    }
}

// ---------------------------------------------------------------------------
// K3: softmax over R + attention-weighted sum over vf (HBM-bound pass:
//     411 MB of vf at 23.3 TB/s => ~18 us floor).
// ---------------------------------------------------------------------------
__global__ void __launch_bounds__(256)
k3_softmax_wsum(const float* __restrict__ vf,
                const float* __restrict__ scores,
                float* __restrict__ out) {
    const int b = blockIdx.x;
    const int tid = threadIdx.x;
    __shared__ float att[NR];
    __shared__ float red[256];

    if (tid < NR) att[tid] = scores[(size_t)b * NR + tid];
    __syncthreads();

    // max
    float m = -INFINITY;
    if (tid < NR) m = att[tid];
    red[tid] = m;
    __syncthreads();
    for (int s = 128; s > 0; s >>= 1) {
        if (tid < s) red[tid] = fmaxf(red[tid], red[tid + s]);
        __syncthreads();
    }
    m = red[0];
    __syncthreads();

    // exp + sum
    float e = 0.0f;
    if (tid < NR) { e = __expf(att[tid] - m); att[tid] = e; }
    red[tid] = e;
    __syncthreads();
    for (int s = 128; s > 0; s >>= 1) {
        if (tid < s) red[tid] += red[tid + s];
        __syncthreads();
    }
    const float inv = 1.0f / red[0];
    __syncthreads();

    // out[b][d] = (1/sum) * sum_r att[r] * vf[b][r][d], coalesced over d.
    // Unroll r by 4 to keep many global_load_b32 in flight (HBM saturation).
    const float* vfb = vf + (size_t)b * NR * DV;
    #pragma unroll
    for (int dd = 0; dd < DV / 256; ++dd) {
        const int d = dd * 256 + tid;
        float acc0 = 0.0f, acc1 = 0.0f, acc2 = 0.0f, acc3 = 0.0f;
        #pragma unroll 1
        for (int r2 = 0; r2 < NR - 3; r2 += 4) {
            acc0 = fmaf(att[r2 + 0], vfb[(size_t)(r2 + 0) * DV + d], acc0);
            acc1 = fmaf(att[r2 + 1], vfb[(size_t)(r2 + 1) * DV + d], acc1);
            acc2 = fmaf(att[r2 + 2], vfb[(size_t)(r2 + 2) * DV + d], acc2);
            acc3 = fmaf(att[r2 + 3], vfb[(size_t)(r2 + 3) * DV + d], acc3);
        }
        // NR = 196 = 49*4, no remainder
        out[(size_t)b * DV + d] = (acc0 + acc1 + acc2 + acc3) * inv;
    }
}

// ---------------------------------------------------------------------------
extern "C" void kernel_launch(void* const* d_in, const int* in_sizes, int n_in,
                              void* d_out, int out_size, void* d_ws, size_t ws_size,
                              hipStream_t stream) {
    const float* vf     = (const float*)d_in[0];
    const float* hidden = (const float*)d_in[1];
    const float* Wv     = (const float*)d_in[2];
    const float* bv     = (const float*)d_in[3];
    const float* Wh     = (const float*)d_in[4];
    const float* bh     = (const float*)d_in[5];
    const float* Wa     = (const float*)d_in[6];
    const float* ba     = (const float*)d_in[7];
    float* out = (float*)d_out;

    // workspace layout
    char* w = (char*)d_ws;
    float*        hp     = (float*)(w);                         // 512*256 f32 = 512 KB
    unsigned int* bhi    = (unsigned int*)(w + (512u << 10));   // 512 KB
    unsigned int* blo    = (unsigned int*)(w + (1024u << 10));  // 512 KB
    float*        scores = (float*)(w + (1536u << 10));         // 512*196 f32

    // K0: swizzle+split Wv into bf16 hi/lo fragment buffers (131072 words)
    k0_swizzle_wv<<<512, 256, 0, stream>>>(Wv, bhi, blo);

    // K1: h_proj
    k1_hproj<<<NB, DA, 0, stream>>>(hidden, Wh, bh, hp);

    // K2: WMMA scores (13 M-tiles x 512 batches, 4 waves each)
    k2_scores<<<dim3(MT, NB), 128, 0, stream>>>(vf, bhi, blo, hp, bv, Wa, ba, scores);

    // K3: softmax + weighted sum
    k3_softmax_wsum<<<NB, 256, 0, stream>>>(vf, scores, out);
}
